// HalfEdgeConv_13666585936438
// MI455X (gfx1250) — matrix operand order repaired and checked by reading, not compiled, not run on hardware
//
#include <hip/hip_runtime.h>
#include <hip/hip_bf16.h>

typedef __attribute__((ext_vector_type(16))) __bf16 v16bf;
typedef __attribute__((ext_vector_type(8)))  float  v8f;
typedef __attribute__((ext_vector_type(4)))  float  f32x4;

#define E_EDGES 262144
#define C_IN    128
#define K_NB    4
#define O_OUT   256
#define KTOT    (K_NB * C_IN)   // 512

#define MTILE   128
#define APAD    136             // A LDS row stride in bf16 (272B): conflict-free ds_b128
#define BTSTRIDE 1056           // B LDS tile stride in bytes (1024 + bank skew room)

// ---------------- prepass 1: x f32 -> bf16 (streaming read, cached write) ----------------
__global__ void cvt_x_kernel(const float* __restrict__ x, __bf16* __restrict__ xb) {
    long t    = (long)blockIdx.x * blockDim.x + threadIdx.x;
    long base = t * 8;
    f32x4 a = __builtin_nontemporal_load(reinterpret_cast<const f32x4*>(x + base));
    f32x4 b = __builtin_nontemporal_load(reinterpret_cast<const f32x4*>(x + base + 4));
    union { __bf16 h[8]; uint4 u; } o;
    o.h[0] = (__bf16)a.x; o.h[1] = (__bf16)a.y; o.h[2] = (__bf16)a.z; o.h[3] = (__bf16)a.w;
    o.h[4] = (__bf16)b.x; o.h[5] = (__bf16)b.y; o.h[6] = (__bf16)b.z; o.h[7] = (__bf16)b.w;
    *reinterpret_cast<uint4*>(xb + base) = o.u;
}

// ---------------- prepass 2: W -> bf16 packed in WMMA B-fragment order -------------------
// [kt=K/32][nt=N/16][lane 0..31][16 bf16]; lane L: N = nt*16+L%16, K = kt*32+(L<16?0:16)+e
__global__ void pack_w_kernel(const float* __restrict__ W, __bf16* __restrict__ Wp) {
    int t    = blockIdx.x * blockDim.x + threadIdx.x;
    int kt   = t >> 13;
    int r    = t & 8191;
    int nt   = r >> 9;
    int r2   = r & 511;
    int lane = r2 >> 4;
    int e    = r2 & 15;
    int k    = kt * 32 + ((lane & 16) ? 16 : 0) + e;
    int n    = nt * 16 + (lane & 15);
    Wp[t] = (__bf16)W[k * O_OUT + n];
}

// ---------------- helpers ----------------------------------------------------------------
__device__ __forceinline__ v16bf load_frag(const __bf16* p) {
    union { uint4 u[2]; v16bf v; } f;
    f.u[0] = *reinterpret_cast<const uint4*>(p);
    f.u[1] = *reinterpret_cast<const uint4*>(p + 8);
    return f.v;
}

// CDNA5 async copy global->LDS (16B per lane), tracked by ASYNCcnt (not LOADcnt)
__device__ __forceinline__ void async_cp16(unsigned lds_off, const void* gptr) {
    asm volatile("global_load_async_to_lds_b128 %0, %1, off"
                 :: "v"(lds_off), "v"(gptr) : "memory");
}
__device__ __forceinline__ void wait_async0() {
    asm volatile("s_wait_asynccnt 0x0" ::: "memory");
}

// ---------------- main kernel: 1024 threads = 32 waves, 16 rows x 64 cols per wave -------
__launch_bounds__(1024, 1)
__global__ void heconv_kernel(const __bf16* __restrict__ xb,
                              const int*    __restrict__ nidx,
                              const __bf16* __restrict__ Wp,
                              const float*  __restrict__ bias,
                              float*        __restrict__ out) {
    __shared__ __align__(16) __bf16 Abuf[2][MTILE * APAD];          // 2 x 34 KB
    __shared__ __align__(16) char   Bbuf[2][32 * BTSTRIDE];         // 2 x 33 KB

    const int tid   = threadIdx.x;
    const int lane  = tid & 31;
    const int wave  = tid >> 5;           // 0..31
    const int wm    = wave & 7;           // 8 row groups of 16
    const int wn    = wave >> 3;          // 4 col groups of 64
    const int mbase = blockIdx.x * MTILE;
    const int hi16  = (lane & 16);
    const int l15   = lane & 15;
    const int n0    = wn * 64;
    const int skew  = (lane >> 3) * 16;   // B bank skew (bytes)

    // neighbor indices for this thread's 2 gather rows, in registers
    int idxv[2][4];
    #pragma unroll
    for (int i = 0; i < 2; ++i) {
        int row = (tid >> 4) + i * 64;
        const int4 q = *reinterpret_cast<const int4*>(nidx + (long)(mbase + row) * 4);
        idxv[i][0] = q.x; idxv[i][1] = q.y; idxv[i][2] = q.z; idxv[i][3] = q.w;
    }

    // async gather of A chunk kb: 128 rows x 256B, 16B chunks, 2/thread
    auto gatherA = [&](int kb, int buf) {
        #pragma unroll
        for (int i = 0; i < 2; ++i) {
            int it  = tid + i * 1024;
            int row = it >> 4;
            int seg = it & 15;
            int e   = idxv[i][kb];        // kb is compile-time after unroll
            unsigned dst = (unsigned)(uintptr_t)&Abuf[buf][row * APAD + seg * 8];
            async_cp16(dst, xb + (long)e * C_IN + seg * 8);
        }
    };

    // async copy of B half-chunk hc (2 kt = 32 tiles = 32KB contiguous in Wp), skewed
    auto copyB = [&](int hc, int buf) {
        const __bf16* src = Wp + hc * 16384;  // elements
        #pragma unroll
        for (int i = 0; i < 2; ++i) {
            int c    = tid + i * 1024;        // 0..2047 chunks of 16B
            int tile = c >> 6;
            int w    = c & 63;
            int ln   = w >> 1;
            int hf   = w & 1;
            unsigned dst = (unsigned)(uintptr_t)
                &Bbuf[buf][tile * BTSTRIDE + ln * 32 + (ln >> 3) * 16 + hf * 16];
            async_cp16(dst, src + tile * 512 + ln * 16 + hf * 8);
        }
    };

    // prologue: first B half-chunk + first A tile in flight
    copyB(0, 0);
    gatherA(0, 0);

    // bias preload (epilogue values; C/D layout: lane -> N = l15)
    float bv[4];
    #pragma unroll
    for (int nt = 0; nt < 4; ++nt) bv[nt] = bias[n0 + nt * 16 + l15];

    // accumulators start at 0 (first WMMA gets inline C=0)
    v8f acc[4];
    #pragma unroll
    for (int i = 0; i < 4; ++i)
        #pragma unroll
        for (int j = 0; j < 8; ++j) acc[i][j] = 0.0f;

    wait_async0();
    __syncthreads();

    #pragma unroll
    for (int kb = 0; kb < 4; ++kb) {
        const __bf16* arow0 = &Abuf[kb & 1][(wm * 16 + l15) * APAD + (hi16 ? 16 : 0)];
        #pragma unroll
        for (int h = 0; h < 2; ++h) {
            const int hc = kb * 2 + h;
            if (hc < 7) copyB(hc + 1, (hc + 1) & 1);          // overlaps with WMMAs
            if (h == 0 && kb < 3) gatherA(kb + 1, (kb + 1) & 1);

            const char* bbase = &Bbuf[hc & 1][wn * 4 * BTSTRIDE + lane * 32 + skew];

            // batch-load both A fragments + first 4 B fragments, then WMMA (staggered waits)
            v16bf a0 = load_frag(arow0 + (h * 2 + 0) * 32);
            v16bf a1 = load_frag(arow0 + (h * 2 + 1) * 32);
            v16bf bf[4];
            #pragma unroll
            for (int j = 0; j < 4; ++j)
                bf[j] = load_frag(reinterpret_cast<const __bf16*>(bbase + j * BTSTRIDE));
            #pragma unroll
            for (int j = 0; j < 4; ++j)
                acc[j] = __builtin_amdgcn_wmma_f32_16x16x32_bf16(
                    false, a0, false, bf[j], (short)0, acc[j], false, false);
            #pragma unroll
            for (int j = 0; j < 4; ++j)
                bf[j] = load_frag(reinterpret_cast<const __bf16*>(bbase + (16 + j) * BTSTRIDE));
            #pragma unroll
            for (int j = 0; j < 4; ++j)
                acc[j] = __builtin_amdgcn_wmma_f32_16x16x32_bf16(
                    false, a1, false, bf[j], (short)0, acc[j], false, false);

            wait_async0();     // next-segment async copies landed in LDS
            __syncthreads();   // visible to all waves; guards buffer reuse
        }
    }

    // epilogue: bias + ReLU + non-temporal f32 stores (keep xb/Wp resident in L2)
    const int mrow = mbase + wm * 16 + (hi16 ? 8 : 0);
    #pragma unroll
    for (int nt = 0; nt < 4; ++nt) {
        int n = n0 + nt * 16 + l15;
        #pragma unroll
        for (int r = 0; r < 8; ++r) {
            float v = acc[nt][r] + bv[nt];
            v = v > 0.0f ? v : 0.0f;
            __builtin_nontemporal_store(v, out + (long)(mrow + r) * O_OUT + n);
        }
    }
}

extern "C" void kernel_launch(void* const* d_in, const int* in_sizes, int n_in,
                              void* d_out, int out_size, void* d_ws, size_t ws_size,
                              hipStream_t stream) {
    const float* x    = (const float*)d_in[0];
    const int*   nidx = (const int*)  d_in[1];
    const float* W    = (const float*)d_in[2];
    const float* b    = (const float*)d_in[3];
    float*       out  = (float*)d_out;

    __bf16* xb = (__bf16*)d_ws;                                           // 64 MB
    __bf16* Wp = (__bf16*)((char*)d_ws + (size_t)E_EDGES * C_IN * 2);     // +256 KB

    cvt_x_kernel <<< (E_EDGES * C_IN / 8) / 256, 256, 0, stream >>>(x, xb);
    pack_w_kernel<<< (KTOT * O_OUT) / 256,       256, 0, stream >>>(W, Wp);
    heconv_kernel<<< E_EDGES / MTILE,            1024, 0, stream >>>(xb, nidx, Wp, b, out);
}
